// DiagonalStateElmanCell_66142496358517
// MI455X (gfx1250) — matrix operand order repaired and checked by reading, not compiled, not run on hardware
//
#include <hip/hip_runtime.h>

// Problem constants (from reference)
#define DM 1024                      // d_model
#define DS 2048                      // d_state
#define TT 2048                      // T
#define BB 8                         // batch
#define MROWS (TT*BB)                // 16384 GEMM rows
static const long Y_SIZE  = (long)MROWS * DM;            // 16,777,216 floats
static const long HALL_OFF = Y_SIZE;                     // h_all starts here in d_out
static const long HS_OFF  = Y_SIZE + (long)BB * DS;      // hs == h_all[1:] (contiguous)

typedef __attribute__((ext_vector_type(8)))  float  v8f;
typedef __attribute__((ext_vector_type(16))) __bf16 v16bf;

__device__ __forceinline__ float fast_exp(float x) {
    return __builtin_amdgcn_exp2f(x * 1.4426950408889634f);
}
__device__ __forceinline__ float fast_sigmoid(float x) {
    return __builtin_amdgcn_rcpf(1.0f + fast_exp(-x));
}
__device__ __forceinline__ float fast_tanh(float x) {
    // tanh(x) = 1 - 2/(e^{2x}+1); saturates correctly for |x| large (exp2->inf/0)
    float e = __builtin_amdgcn_exp2f(x * 2.8853900817779268f); // 2*log2(e)
    return 1.0f - 2.0f * __builtin_amdgcn_rcpf(e + 1.0f);
}

// ---------------------------------------------------------------------------
// Pack a row-major f32 [M,K] matrix into the WMMA A-fragment lane layout,
// split into bf16 hi/lo planes. Linear packed index:
//   gid = (((mtile*Ktiles + ktile)*32 + lane)*16 + j)
// A 16x32 k order used here:  k = ktile*32 + 16*(lane>>4) + j,
//                             m = mtile*16 + (lane&15)
// ---------------------------------------------------------------------------
__global__ __launch_bounds__(256) void pack_a_kernel(
    const float* __restrict__ src, __bf16* __restrict__ hi,
    __bf16* __restrict__ lo, int K)
{
    long gid  = (long)blockIdx.x * 256 + threadIdx.x;
    int  j    = (int)(gid & 15);
    int  lane = (int)((gid >> 4) & 31);
    long rest = gid >> 9;
    int  Kt   = K >> 5;
    int  kt   = (int)(rest % Kt);
    long mt   = rest / Kt;
    long m    = mt * 16 + (lane & 15);
    int  k    = kt * 32 + ((lane >> 4) << 4) + j;
    float v   = src[m * (long)K + k];
    __bf16 h  = (__bf16)v;
    hi[gid] = h;
    lo[gid] = (__bf16)(v - (float)h);
}

// ---------------------------------------------------------------------------
// Pack a row-major f32 [K,N] matrix into the WMMA B-fragment lane layout.
//   gid = (((ntile*Ktiles + ktile)*32 + lane)*16 + j)
// B 32x16 order used here: lanes0-15 hold K=0..15, lanes16-31 K=16..31 ->
//   k = ktile*32 + 16*(lane>>4) + j,   n = ntile*16 + (lane&15)
// ---------------------------------------------------------------------------
__global__ __launch_bounds__(256) void pack_b_kernel(
    const float* __restrict__ src, __bf16* __restrict__ hi,
    __bf16* __restrict__ lo, int K, int N)
{
    long gid  = (long)blockIdx.x * 256 + threadIdx.x;
    int  j    = (int)(gid & 15);
    int  lane = (int)((gid >> 4) & 31);
    long rest = gid >> 9;
    int  Kt   = K >> 5;
    int  kt   = (int)(rest % Kt);
    long nt   = rest / Kt;
    long n    = nt * 16 + (lane & 15);
    int  k    = kt * 32 + ((lane >> 4) << 4) + j;
    float v   = src[(long)k * N + n];
    __bf16 h  = (__bf16)v;
    hi[gid] = h;
    lo[gid] = (__bf16)(v - (float)h);
}

// ---------------------------------------------------------------------------
// WMMA GEMM: Out[M,N] = A[M,K] x B[K,N], bf16x3 split-precision
// (a_hi*b_hi + a_hi*b_lo + a_lo*b_hi accumulated in f32).
// Each wave owns a 64(M) x 16(N) tile: 4 accumulators, B fragment reused 4x.
// EPI==1: Out *= silu(Z) elementwise (Z same [M,N] layout).
// ---------------------------------------------------------------------------
template<int EPI>
__global__ __launch_bounds__(256) void gemm_wmma_kernel(
    const __bf16* __restrict__ Ah, const __bf16* __restrict__ Al,
    const __bf16* __restrict__ Bh, const __bf16* __restrict__ Bl,
    const float*  __restrict__ Z,  float* __restrict__ Out,
    int Ntiles, int Ktiles, int N)
{
    int wave = blockIdx.x * 8 + (threadIdx.x >> 5);
    int lane = threadIdx.x & 31;
    int nt = wave % Ntiles;
    int mg = wave / Ntiles;           // 64-row macro-tile index

    const v16bf* AhV = (const v16bf*)Ah;
    const v16bf* AlV = (const v16bf*)Al;
    const v16bf* BhV = (const v16bf*)Bh;
    const v16bf* BlV = (const v16bf*)Bl;

    long ms   = (long)Ktiles * 32;           // v16bf stride per 16-row A tile
    long aoff = (long)(mg * 4) * ms + lane;
    long boff = (long)nt * ms + lane;

    v8f acc[4] = {};

    for (int kt = 0; kt < Ktiles; ++kt) {
        long bo = boff + (long)kt * 32;
        v16bf bh = BhV[bo];
        v16bf bl = BlV[bo];
        long ao = aoff + (long)kt * 32;
#pragma unroll
        for (int mt = 0; mt < 4; ++mt) {
            v16bf ah = AhV[ao + mt * ms];
            v16bf al = AlV[ao + mt * ms];
            acc[mt] = __builtin_amdgcn_wmma_f32_16x16x32_bf16(
                false, ah, false, bh, (short)0, acc[mt], false, false);
            acc[mt] = __builtin_amdgcn_wmma_f32_16x16x32_bf16(
                false, ah, false, bl, (short)0, acc[mt], false, false);
            acc[mt] = __builtin_amdgcn_wmma_f32_16x16x32_bf16(
                false, al, false, bh, (short)0, acc[mt], false, false);
        }
    }

    // C/D layout: VGPR r -> row r + 8*(lane>>4), col = lane&15
    int hlf = lane >> 4, col = lane & 15;
    long n = (long)nt * 16 + col;
#pragma unroll
    for (int mt = 0; mt < 4; ++mt) {
        long mbase = (long)(mg * 4 + mt) * 16 + 8 * hlf;
#pragma unroll
        for (int r = 0; r < 8; ++r) {
            long o = (mbase + r) * (long)N + n;
            float v = acc[mt][r];
            if (EPI) {
                float zz = Z[o];
                v *= zz * fast_sigmoid(zz);
            }
            Out[o] = v;
        }
    }
}

// ---------------------------------------------------------------------------
// Sequential scan, in place: buf holds Bx[t] (written by GEMM1 into the
// h_all[t+1] slot); each thread owns one (b, n_state) chain and rewrites
// buf[t] with h[t]. 8-deep batched prefetch hides load latency (mostly L2).
// ---------------------------------------------------------------------------
__global__ __launch_bounds__(256) void scan_kernel(
    const float* __restrict__ h0, const float* __restrict__ logA,
    float* __restrict__ hall)
{
    int idx = blockIdx.x * 256 + threadIdx.x;   // 0 .. B*DS-1
    int n = idx & (DS - 1);
    float A = fast_sigmoid(logA[n]);
    float h = h0[idx];
    hall[idx] = h;                              // h_all[0] = h0

    float* buf = hall + BB * DS;                // hs region (aliases Bx)
    const int S = BB * DS;

    float cur[8];
#pragma unroll
    for (int u = 0; u < 8; ++u) cur[u] = buf[(long)u * S + idx];

    for (int tb = 0; tb < TT; tb += 8) {
        float nxt[8];
        bool more = (tb + 8) < TT;
#pragma unroll
        for (int u = 0; u < 8; ++u)
            nxt[u] = more ? buf[(long)(tb + 8 + u) * S + idx] : 0.0f;
#pragma unroll
        for (int u = 0; u < 8; ++u) {
            h = fast_tanh(fmaf(A, h, cur[u]));
            buf[(long)(tb + u) * S + idx] = h;
        }
#pragma unroll
        for (int u = 0; u < 8; ++u) cur[u] = nxt[u];
    }
}

// ---------------------------------------------------------------------------
// Launch: pack B,C,x -> GEMM1 (Bx into h_all[1:] slot) -> in-place scan ->
// pack hs -> GEMM2 with silu(z) epilogue into y.
// Workspace layout (bf16 elements): Bpack hi/lo (2x2M), Cpack hi/lo (2x2M),
// then a 134 MB A-pack region reused for x-pack (67 MB) and later hs-pack.
// Total ws use ~150 MB.
// ---------------------------------------------------------------------------
extern "C" void kernel_launch(void* const* d_in, const int* in_sizes, int n_in,
                              void* d_out, int out_size, void* d_ws, size_t ws_size,
                              hipStream_t stream)
{
    const float* x    = (const float*)d_in[0];
    const float* z    = (const float*)d_in[1];
    const float* h0   = (const float*)d_in[2];
    const float* Bm   = (const float*)d_in[3];
    const float* Cm   = (const float*)d_in[4];
    const float* logA = (const float*)d_in[5];
    float* out = (float*)d_out;

    __bf16* ws = (__bf16*)d_ws;
    const long WPLANE = (long)DM * DS;          // 2M elements per weight plane
    __bf16* Bh = ws;
    __bf16* Bl = Bh + WPLANE;
    __bf16* Ch = Bl + WPLANE;
    __bf16* Cl = Ch + WPLANE;
    __bf16* Ap = Cl + WPLANE;                   // reusable A-pack region
    __bf16* Xh = Ap;
    __bf16* Xl = Xh + (long)MROWS * DM;
    __bf16* Hh = Ap;                            // reuses x-pack space (x dead)
    __bf16* Hl = Hh + (long)MROWS * DS;

    // 1) pack weights (one-time, small)
    pack_b_kernel<<<(int)(((long)DM * DS) / 256), 256, 0, stream>>>(Bm, Bh, Bl, DM, DS);
    pack_b_kernel<<<(int)(((long)DS * DM) / 256), 256, 0, stream>>>(Cm, Ch, Cl, DS, DM);

    // 2) pack x into A-fragment layout (bf16 hi/lo)
    pack_a_kernel<<<(int)(((long)MROWS * DM) / 256), 256, 0, stream>>>(x, Xh, Xl, DM);

    // 3) GEMM1: Bx = x @ B, written directly into h_all[1:] slots of d_out
    {
        int Ntiles = DS / 16, Ktiles = DM / 32;
        int waves = (MROWS / 64) * Ntiles;      // 32768 waves -> 4096 blocks
        gemm_wmma_kernel<0><<<waves / 8, 256, 0, stream>>>(
            Xh, Xl, Bh, Bl, nullptr, out + HS_OFF, Ntiles, Ktiles, DS);
    }

    // 4) recurrence, transforming Bx -> hs in place; also writes h_all[0]
    scan_kernel<<<(BB * DS) / 256, 256, 0, stream>>>(h0, logA, out + HALL_OFF);

    // 5) pack hs into A-fragment layout
    pack_a_kernel<<<(int)(((long)MROWS * DS) / 256), 256, 0, stream>>>(out + HS_OFF, Hh, Hl, DS);

    // 6) GEMM2: y = hs @ C, fused with y *= silu(z)
    {
        int Ntiles = DM / 16, Ktiles = DS / 32;
        int waves = (MROWS / 64) * Ntiles;      // 16384 waves -> 2048 blocks
        gemm_wmma_kernel<1><<<waves / 8, 256, 0, stream>>>(
            Hh, Hl, Ch, Cl, z, out, Ntiles, Ktiles, DM);
    }
}
